// RoundEncoder_74191265071512
// MI455X (gfx1250) — compile-verified, matrix operand order checked by reading, not run on hardware
//
#include <hip/hip_runtime.h>
#include <math.h>

// ---------------- problem constants ----------------
#define B_SZ     512
#define S_TOK    13
#define M_TOK    (B_SZ * S_TOK)      // 6656 tokens
#define D_MODEL  1024
#define DH_HALF  512
#define H_HEADS  16
#define HD_DIM   64
#define DFF_DIM  4096
#define L_LAYERS 4
#define EPS_LN   1e-5f

typedef __bf16 bf16_t;
typedef __attribute__((ext_vector_type(16))) __bf16 v16bf;
typedef __attribute__((ext_vector_type(8)))  float  v8f;
typedef __attribute__((ext_vector_type(4)))  int    v4i_t;

// ---------------- CDNA5 async global->LDS staging (guarded) ----------------
#if defined(__has_builtin)
#  if __has_builtin(__builtin_amdgcn_global_load_async_to_lds_b128)
#    define HAVE_ASYNC_LDS 1
#  endif
#  if __has_builtin(__builtin_amdgcn_s_wait_asynccnt)
#    define HAVE_WAIT_ASYNC_BUILTIN 1
#  endif
#endif

#ifdef HAVE_ASYNC_LDS
typedef __attribute__((address_space(1))) v4i_t* as1_v4i_p;
typedef __attribute__((address_space(3))) v4i_t* as3_v4i_p;
#endif

__device__ __forceinline__ void copy16_g2l(const bf16_t* __restrict__ g, bf16_t* l) {
#ifdef HAVE_ASYNC_LDS
    // generic global ptr is identity-mapped to AS1; generic shared ptr's low 32
    // bits are the AS3 offset (aperture_hi<<32 | lds_offset).
    __builtin_amdgcn_global_load_async_to_lds_b128(
        (as1_v4i_p)(uintptr_t)g, (as3_v4i_p)(uintptr_t)l, 0, 0);
#else
    *(uint4*)l = *(const uint4*)g;   // sync fallback: global->VGPR->ds_store
#endif
}

template <int N>
__device__ __forceinline__ void wait_async() {
#ifdef HAVE_ASYNC_LDS
#  ifdef HAVE_WAIT_ASYNC_BUILTIN
    __builtin_amdgcn_s_wait_asynccnt(N);
#  else
    asm volatile("s_wait_asynccnt %0" ::"n"(N));
#  endif
#endif
}

// ---------------- f32 -> bf16 conversion ----------------
__global__ __launch_bounds__(256) void cvt_bf16_kernel(const float* __restrict__ src,
                                                       bf16_t* __restrict__ dst, int n) {
    int i = blockIdx.x * 256 + threadIdx.x;
    if (i < n) dst[i] = (bf16_t)src[i];
}

// ---------------- embedding / thermometer encode ----------------
__global__ __launch_bounds__(256) void embed_kernel(const int* __restrict__ sparse,
                                                    const int* __restrict__ numeric,
                                                    const float* __restrict__ semb,
                                                    const float* __restrict__ nemb,
                                                    float* __restrict__ x,
                                                    bf16_t* __restrict__ xbf) {
    const int r = blockIdx.x;
    const int b = r / S_TOK, tk = r % S_TOK;
    const int t = threadIdx.x;
#pragma unroll
    for (int i = 0; i < 4; ++i) {
        const int d = t * 4 + i;
        float val;
        if (tk < 7) {
            const int idx = sparse[b * 7 + tk];
            val = semb[(size_t)idx * D_MODEL + d];
        } else {
            const int j = tk - 7;
            if (d < DH_HALF) {
                const float scale = (j < 2) ? 1.0f : (512.0f / 100000.0f); // DH/maxs[j]
                const float scaled = (float)numeric[b * 6 + j] * scale;
                val = fminf(fmaxf(scaled - (float)d, 0.0f), 1.0f);
            } else {
                val = nemb[j * DH_HALF + (d - DH_HALF)];
            }
        }
        x[(size_t)r * D_MODEL + d] = val;
        xbf[(size_t)r * D_MODEL + d] = (bf16_t)val;
    }
}

// ---------------- WMMA GEMM:  C[M,N] = act(A[M,K] * W[N,K]^T + bias[N]) ----------------
// block: 256 threads (8 wave32), 128x128 output tile; wave -> 32(M) x 64(N) sub-tile
// (2 A-frags x 4 B-frags = 8 WMMA / K-step). Double-buffered LDS 128x32 bf16 per operand,
// staged via GLOBAL_LOAD_ASYNC_TO_LDS_B128 when available.
#define BM 128
#define BN 128
#define BK 32

template <int GELU, int OUTF32>
__global__ __launch_bounds__(256) void gemm_wmma(const bf16_t* __restrict__ A,
                                                 const bf16_t* __restrict__ W,
                                                 const float* __restrict__ bias,
                                                 float* __restrict__ Cf,
                                                 bf16_t* __restrict__ Cb,
                                                 int Ndim, int Kdim) {
    __shared__ __align__(16) bf16_t lA[2][BM * BK];
    __shared__ __align__(16) bf16_t lB[2][BN * BK];

    const int t = threadIdx.x;
    const int m0 = blockIdx.y * BM, n0 = blockIdx.x * BN;

    const int lane = t & 31, wave = t >> 5;
    const int wm = wave >> 1;          // 0..3 -> rows wm*32
    const int wn = wave & 1;           // 0..1 -> cols wn*64
    const int lrow  = lane & 15;
    const int khalf = lane >> 4;       // ISA 16-bit layout: lanes16-31 carry K+8
    const int kb    = khalf * 8;

    v8f acc[2][4];
    const v8f vzero = {0.f, 0.f, 0.f, 0.f, 0.f, 0.f, 0.f, 0.f};
#pragma unroll
    for (int mi = 0; mi < 2; ++mi)
#pragma unroll
        for (int ni = 0; ni < 4; ++ni) acc[mi][ni] = vzero;

    const int nK = Kdim >> 5;

    auto stage_load = [&](int s, int kt) {
        const int k0 = kt << 5;
#pragma unroll
        for (int i = 0; i < 2; ++i) {
            const int c = t + 256 * i;           // 512 chunks of 16B per operand
            const int row = c >> 2;              // 0..127
            const int kc  = (c & 3) << 3;        // 0,8,16,24
            copy16_g2l(A + (size_t)(m0 + row) * Kdim + k0 + kc, &lA[s][row * BK + kc]);
            copy16_g2l(W + (size_t)(n0 + row) * Kdim + k0 + kc, &lB[s][row * BK + kc]);
        }
    };

    stage_load(0, 0);

    for (int kt = 0; kt < nK; ++kt) {
        const int s = kt & 1;
        if (kt + 1 < nK) {
            stage_load(s ^ 1, kt + 1);           // issue next stage (4 async ops/thread)
            wait_async<4>();                     // in-order: stage s ops now complete
            if (kt + 2 < nK) {                   // L2 prefetch of tile after next
                const int kp = (kt + 2) << 5;
                __builtin_prefetch(A + (size_t)(m0 + (t >> 1)) * Kdim + kp, 0, 1);
                __builtin_prefetch(W + (size_t)(n0 + (t >> 1)) * Kdim + kp, 0, 1);
            }
        } else {
            wait_async<0>();
        }
        __syncthreads();                          // stage s visible to all waves

        union { v16bf v; uint4 u[2]; } af[2], bfr[4];
#pragma unroll
        for (int mi = 0; mi < 2; ++mi) {
            const bf16_t* p = &lA[s][(wm * 32 + mi * 16 + lrow) * BK + kb];
            af[mi].u[0] = *(const uint4*)p;       // K kb..kb+7
            af[mi].u[1] = *(const uint4*)(p + 16);// K kb+16..kb+23
        }
#pragma unroll
        for (int ni = 0; ni < 4; ++ni) {
            const bf16_t* p = &lB[s][(wn * 64 + ni * 16 + lrow) * BK + kb];
            bfr[ni].u[0] = *(const uint4*)p;
            bfr[ni].u[1] = *(const uint4*)(p + 16);
        }
#pragma unroll
        for (int mi = 0; mi < 2; ++mi)
#pragma unroll
            for (int ni = 0; ni < 4; ++ni)
                acc[mi][ni] = __builtin_amdgcn_wmma_f32_16x16x32_bf16(
                    false, af[mi].v, false, bfr[ni].v, (short)0, acc[mi][ni], false, false);

        __syncthreads();                          // protect stage s before next overwrite
    }

    // epilogue: C layout -> reg r: lanes<16 M=r, lanes>=16 M=r+8; N = lane&15
    const int gmBase = m0 + wm * 32 + khalf * 8;
#pragma unroll
    for (int mi = 0; mi < 2; ++mi) {
#pragma unroll
        for (int ni = 0; ni < 4; ++ni) {
            const int gn = n0 + wn * 64 + ni * 16 + lrow;
            const float bb = bias[gn];
#pragma unroll
            for (int r = 0; r < 8; ++r) {
                float v = acc[mi][ni][r] + bb;
                if (GELU) v = 0.5f * v * (1.0f + erff(v * 0.70710678118f));
                const size_t o = (size_t)(gmBase + mi * 16 + r) * Ndim + gn;
                if (OUTF32) Cf[o] = v;
                else        Cb[o] = (bf16_t)v;
            }
        }
    }
}

// ---------------- attention (S=13, HD=64), one block per batch ----------------
__global__ __launch_bounds__(256) void attn_kernel(const float* __restrict__ qkv,
                                                   bf16_t* __restrict__ obf) {
    const int b = blockIdx.x;
    const int t = threadIdx.x;
    if (t >= H_HEADS * S_TOK) return;
    const int h = t / S_TOK, q = t % S_TOK;
    const float* base = qkv + (size_t)b * S_TOK * 3 * D_MODEL;
    const float* qv = base + (size_t)q * 3 * D_MODEL + h * HD_DIM;

    float qreg[HD_DIM];
#pragma unroll
    for (int d = 0; d < HD_DIM; ++d) qreg[d] = qv[d];

    float sc[S_TOK];
    float mx = -1e30f;
    for (int k = 0; k < S_TOK; ++k) {
        const float* kv = base + (size_t)k * 3 * D_MODEL + D_MODEL + h * HD_DIM;
        float a = 0.0f;
#pragma unroll
        for (int d = 0; d < HD_DIM; ++d) a += qreg[d] * kv[d];
        a *= 0.125f;                   // 1/sqrt(64)
        sc[k] = a;
        mx = fmaxf(mx, a);
    }
    float denom = 0.0f;
    for (int k = 0; k < S_TOK; ++k) { sc[k] = __expf(sc[k] - mx); denom += sc[k]; }
    const float inv = 1.0f / denom;

    float out[HD_DIM];
#pragma unroll
    for (int d = 0; d < HD_DIM; ++d) out[d] = 0.0f;
    for (int k = 0; k < S_TOK; ++k) {
        const float* vv = base + (size_t)k * 3 * D_MODEL + 2 * D_MODEL + h * HD_DIM;
        const float p = sc[k] * inv;
#pragma unroll
        for (int d = 0; d < HD_DIM; ++d) out[d] += p * vv[d];
    }
    bf16_t* op = obf + ((size_t)b * S_TOK + q) * D_MODEL + h * HD_DIM;
#pragma unroll
    for (int d = 0; d < HD_DIM; ++d) op[d] = (bf16_t)out[d];
}

// ---------------- residual + LayerNorm ----------------
__global__ __launch_bounds__(256) void add_ln_kernel(const float* __restrict__ xin,
                                                     const float* __restrict__ delta,
                                                     const float* __restrict__ g,
                                                     const float* __restrict__ bta,
                                                     float* __restrict__ xout,
                                                     bf16_t* __restrict__ xbf) {
    __shared__ float red[16];
    const int r = blockIdx.x;
    const int t = threadIdx.x;
    const float* xr = xin + (size_t)r * D_MODEL;
    const float* dr = delta + (size_t)r * D_MODEL;

    float v[4];
    float s = 0.0f, s2 = 0.0f;
#pragma unroll
    for (int i = 0; i < 4; ++i) {
        const int d = t * 4 + i;
        v[i] = xr[d] + dr[d];
        s += v[i];
        s2 += v[i] * v[i];
    }
    for (int off = 16; off >= 1; off >>= 1) {     // wave32 reduce
        s  += __shfl_xor(s, off, 32);
        s2 += __shfl_xor(s2, off, 32);
    }
    const int wave = t >> 5, lane = t & 31;
    if (lane == 0) { red[wave] = s; red[8 + wave] = s2; }
    __syncthreads();
    if (t == 0) {
        float a = 0.0f, c = 0.0f;
        for (int w = 0; w < 8; ++w) { a += red[w]; c += red[8 + w]; }
        red[0] = a; red[8] = c;
    }
    __syncthreads();
    const float mean = red[0] * (1.0f / D_MODEL);
    const float var  = red[8] * (1.0f / D_MODEL) - mean * mean;
    const float rstd = rsqrtf(var + EPS_LN);
#pragma unroll
    for (int i = 0; i < 4; ++i) {
        const int d = t * 4 + i;
        const float y = (v[i] - mean) * rstd * g[d] + bta[d];
        xout[(size_t)r * D_MODEL + d] = y;
        xbf[(size_t)r * D_MODEL + d] = (bf16_t)y;
    }
}

// ---------------- host-side orchestration ----------------
extern "C" void kernel_launch(void* const* d_in, const int* in_sizes, int n_in,
                              void* d_out, int out_size, void* d_ws, size_t ws_size,
                              hipStream_t stream) {
    (void)in_sizes; (void)n_in; (void)out_size; (void)ws_size;
    const int*   sparse  = (const int*)d_in[0];
    const int*   numeric = (const int*)d_in[1];
    const float* semb    = (const float*)d_in[2];
    const float* nemb    = (const float*)d_in[3];
    const float* Wqkv    = (const float*)d_in[4];
    const float* bqkv    = (const float*)d_in[5];
    const float* Wo      = (const float*)d_in[6];
    const float* bo      = (const float*)d_in[7];
    const float* W1      = (const float*)d_in[8];
    const float* b1      = (const float*)d_in[9];
    const float* W2      = (const float*)d_in[10];
    const float* b2      = (const float*)d_in[11];
    const float* ln1_g   = (const float*)d_in[12];
    const float* ln1_b   = (const float*)d_in[13];
    const float* ln2_g   = (const float*)d_in[14];
    const float* ln2_b   = (const float*)d_in[15];

    char* ws = (char*)d_ws;
    size_t off = 0;
    auto carve = [&](size_t bytes) -> void* {
        void* p = ws + off;
        off = (off + bytes + 255) & ~(size_t)255;
        return p;
    };
    float*  x    = (float*)carve((size_t)M_TOK * D_MODEL * 4);
    bf16_t* xbf  = (bf16_t*)carve((size_t)M_TOK * D_MODEL * 2);
    float*  qkv  = (float*)carve((size_t)M_TOK * 3 * D_MODEL * 4);
    bf16_t* obf  = (bf16_t*)carve((size_t)M_TOK * D_MODEL * 2);
    float*  tmp  = (float*)carve((size_t)M_TOK * D_MODEL * 4);
    bf16_t* hbf  = (bf16_t*)carve((size_t)M_TOK * DFF_DIM * 2);
    bf16_t* wbuf = (bf16_t*)carve((size_t)DFF_DIM * D_MODEL * 2);

    embed_kernel<<<M_TOK, 256, 0, stream>>>(sparse, numeric, semb, nemb, x, xbf);

    for (int l = 0; l < L_LAYERS; ++l) {
        const float* Wqkv_l = Wqkv + (size_t)l * 3 * D_MODEL * D_MODEL;
        const float* Wo_l   = Wo   + (size_t)l * D_MODEL * D_MODEL;
        const float* W1_l   = W1   + (size_t)l * DFF_DIM * D_MODEL;
        const float* W2_l   = W2   + (size_t)l * D_MODEL * DFF_DIM;

        // qkv = x @ Wqkv^T + bqkv
        {
            const int n = 3 * D_MODEL * D_MODEL;
            cvt_bf16_kernel<<<(n + 255) / 256, 256, 0, stream>>>(Wqkv_l, wbuf, n);
            dim3 grid(3 * D_MODEL / BN, M_TOK / BM);
            gemm_wmma<0, 1><<<grid, 256, 0, stream>>>(xbf, wbuf, bqkv + (size_t)l * 3 * D_MODEL,
                                                      qkv, nullptr, 3 * D_MODEL, D_MODEL);
        }
        attn_kernel<<<B_SZ, 256, 0, stream>>>(qkv, obf);
        // tmp = o @ Wo^T + bo
        {
            const int n = D_MODEL * D_MODEL;
            cvt_bf16_kernel<<<(n + 255) / 256, 256, 0, stream>>>(Wo_l, wbuf, n);
            dim3 grid(D_MODEL / BN, M_TOK / BM);
            gemm_wmma<0, 1><<<grid, 256, 0, stream>>>(obf, wbuf, bo + (size_t)l * D_MODEL,
                                                      tmp, nullptr, D_MODEL, D_MODEL);
        }
        add_ln_kernel<<<M_TOK, 256, 0, stream>>>(x, tmp, ln1_g + (size_t)l * D_MODEL,
                                                 ln1_b + (size_t)l * D_MODEL, x, xbf);
        // h = gelu(x @ W1^T + b1) (bf16 out)
        {
            const int n = DFF_DIM * D_MODEL;
            cvt_bf16_kernel<<<(n + 255) / 256, 256, 0, stream>>>(W1_l, wbuf, n);
            dim3 grid(DFF_DIM / BN, M_TOK / BM);
            gemm_wmma<1, 0><<<grid, 256, 0, stream>>>(xbf, wbuf, b1 + (size_t)l * DFF_DIM,
                                                      nullptr, hbf, DFF_DIM, D_MODEL);
        }
        // tmp = h @ W2^T + b2
        {
            const int n = D_MODEL * DFF_DIM;
            cvt_bf16_kernel<<<(n + 255) / 256, 256, 0, stream>>>(W2_l, wbuf, n);
            dim3 grid(D_MODEL / BN, M_TOK / BM);
            gemm_wmma<0, 1><<<grid, 256, 0, stream>>>(hbf, wbuf, b2 + (size_t)l * D_MODEL,
                                                      tmp, nullptr, D_MODEL, DFF_DIM);
        }
        float* xo = (l == L_LAYERS - 1) ? (float*)d_out : x;
        add_ln_kernel<<<M_TOK, 256, 0, stream>>>(x, tmp, ln2_g + (size_t)l * D_MODEL,
                                                 ln2_b + (size_t)l * D_MODEL, xo, xbf);
    }
}
